// QuantizationLayer_71098888618379
// MI455X (gfx1250) — compile-verified, matrix operand order checked by reading, not run on hardware
//
#include <hip/hip_runtime.h>
#include <hip/hip_bf16.h>

typedef __attribute__((ext_vector_type(16))) _Float16 v16h;
typedef __attribute__((ext_vector_type(8)))  float    v8f;

#define HH 256
#define WW 256
#define SS 16
#define BB 8
#define H2 128
#define W2 128
#define HW   (HH*WW)    /* 65536 */
#define HW4  (H2*W2)    /* 16384 */

// ---- workspace layout (bytes, 256-aligned regions) ----
#define OFF_TMAX      0u                                   /* 8  x u32  */
#define OFF_CNT       256u                                 /* 128x u32  */
#define OFF_XSUM      1024u
#define OFF_YSUM      1536u
#define OFF_PARAMS    2048u                                /* 128x int4 */
#define OFF_COUNTER4  4096u                                /* B*HW4 i32 */
#define OFF_TIMER4    (OFF_COUNTER4 + BB*HW4*4u)
#define OFF_CONTAINER (OFF_TIMER4   + BB*HW4*4u)           /* B*HW i32  */
#define OFF_COMBINER  (OFF_CONTAINER+ BB*HW*4u)            /* B*S*HW i32*/
#define OFF_SEG_A     (OFF_COMBINER + (size_t)BB*SS*HW*4u)
#define OFF_SEG_B     (OFF_SEG_A    + (size_t)BB*HW*4u)
#define ZERO_BYTES    OFF_SEG_A                            /* everything scattered into */

// ------------------------------------------------------------------ zero
__global__ void k_zero(uint4* __restrict__ p, long n16) {
  long i = (long)blockIdx.x * blockDim.x + threadIdx.x;
  long stride = (long)gridDim.x * blockDim.x;
  uint4 z = {0u, 0u, 0u, 0u};
  for (; i < n16; i += stride) p[i] = z;
}

// ------------------------------------------------------------------ per-batch t_max
__global__ void k_tmax(const float* __restrict__ ev, int n, unsigned* __restrict__ tmaxU) {
  __shared__ unsigned lmax[BB];
  if (threadIdx.x < BB) lmax[threadIdx.x] = 0u;
  __syncthreads();
  long i = (long)blockIdx.x * blockDim.x + threadIdx.x;
  long stride = (long)gridDim.x * blockDim.x;
  for (; i < n; i += stride) {
    float t = ev[i * 5 + 2];                 // t > 0 so float bits order == uint order
    int   b = (int)ev[i * 5 + 4];
    atomicMax(&lmax[b], __float_as_uint(t));
  }
  __syncthreads();
  if (threadIdx.x < BB && lmax[threadIdx.x]) atomicMax(&tmaxU[threadIdx.x], lmax[threadIdx.x]);
}

// ------------------------------------------------------------------ main event scatter
__global__ void k_scatter(const float* __restrict__ ev, int n,
                          const unsigned* __restrict__ tmaxU,
                          int* __restrict__ container, int* __restrict__ combiner,
                          unsigned* __restrict__ cnt, unsigned* __restrict__ xsum,
                          unsigned* __restrict__ ysum,
                          int* __restrict__ counter4, float* __restrict__ timer4) {
  __shared__ unsigned lc[BB * SS], lx[BB * SS], ly[BB * SS];
  if (threadIdx.x < BB * SS) { lc[threadIdx.x] = 0u; lx[threadIdx.x] = 0u; ly[threadIdx.x] = 0u; }
  __syncthreads();
  long i = (long)blockIdx.x * blockDim.x + threadIdx.x;
  long stride = (long)gridDim.x * blockDim.x;
  for (; i < n; i += stride) {
    float x = ev[i * 5 + 0], y = ev[i * 5 + 1], t = ev[i * 5 + 2], bf = ev[i * 5 + 4];
    int b = (int)bf, xi = (int)x, yi = (int)y;
    float tn = t / __uint_as_float(tmaxU[b]);        // normalized time
    int ts = (int)(tn * 16.0f);                      // == searchsorted(arange(1,16)/16, tn, 'right')
    ts = ts > (SS - 1) ? (SS - 1) : ts;
    atomicAdd(&container[(b << 16) + (yi << 8) + xi], 1);
    if (ts) atomicAdd(&combiner[((b * SS + ts) << 16) + (yi << 8) + xi], ts);
    int s = b * SS + ts;                             // LDS-staged segment stats (hot buckets)
    atomicAdd(&lc[s], 1u);
    atomicAdd(&lx[s], (unsigned)xi);
    atomicAdd(&ly[s], (unsigned)yi);
    int q = (b << 14) + ((yi >> 1) << 7) + (xi >> 1);
    atomicAdd(&counter4[q], 1);
    atomicAdd(&timer4[q], tn);
  }
  __syncthreads();
  if (threadIdx.x < BB * SS) {
    unsigned c = lc[threadIdx.x];
    if (c) {
      atomicAdd(&cnt[threadIdx.x], c);
      atomicAdd(&xsum[threadIdx.x], lx[threadIdx.x]);
      atomicAdd(&ysum[threadIdx.x], ly[threadIdx.x]);
    }
  }
}

// ------------------------------------------------------------------ sequential shift recurrence (scalars)
__global__ void k_params(const unsigned* __restrict__ cnt, const unsigned* __restrict__ xsum,
                         const unsigned* __restrict__ ysum, int4* __restrict__ params) {
  int b = threadIdx.x;
  if (b >= BB) return;
  unsigned pts = cnt[b * SS];
  float xm = (float)xsum[b * SS] / (float)pts;
  float ym = (float)ysum[b * SS] / (float)pts;
  for (int i = 1; i < SS; ++i) {
    unsigned ci = cnt[b * SS + i];
    int cond = ci > pts;
    float xmi = (float)xsum[b * SS + i] / (float)ci;
    float ymi = (float)ysum[b * SS + i] / (float)ci;
    float dxf = cond ? (xmi - xm) : (xm - xmi);
    float dyf = cond ? (ymi - ym) : (ym - ymi);
    int dx = (int)floorf(dxf);
    int dy = (int)floorf(dyf);
    if (cond) { xm = xmi; ym = ymi; }
    params[b * SS + i] = make_int4(cond, dy > 0 ? dy : 0, dx > 0 ? dx : 0, 0);
  }
}

// ------------------------------------------------------------------ segment = combiner[:,0]
__global__ void k_seg_init(const int* __restrict__ combiner, float* __restrict__ seg) {
  int gid = blockIdx.x * blockDim.x + threadIdx.x;
  if (gid >= BB * HW) return;
  int b = gid >> 16, rem = gid & (HW - 1);
  seg[gid] = (float)combiner[((b * SS) << 16) + rem];
}

// ------------------------------------------------------------------ one shift+add step
__global__ void k_combine(const int4* __restrict__ params, const int* __restrict__ combiner,
                          const float* __restrict__ src, float* __restrict__ dst, int step) {
  int gid = blockIdx.x * blockDim.x + threadIdx.x;
  if (gid >= BB * HW) return;
  int b = gid >> 16, rem = gid & (HW - 1);
  int y = rem >> 8, x = rem & 255;
  int4 pr = params[b * SS + step];
  const int* comb = combiner + ((long)(b * SS + step) << 16);
  bool inr = (y >= pr.y) && (x >= pr.z);
  int shidx = inr ? (((y - pr.y) << 8) + (x - pr.z)) : rem;  // clamped: always in-bounds
  float res;
  if (pr.x) {  // larger slice: base = combiner[i], tmp = shifted old segment
    res = (float)comb[rem] + (inr ? src[(b << 16) + shidx] : 0.0f);
  } else {     // base = old segment, tmp = shifted combiner[i]
    res = src[gid] + (inr ? (float)comb[shidx] : 0.0f);
  }
  dst[gid] = res;
}

// ------------------------------------------------------------------ WMMA 2x2 +/-1 pooling:
//   dyv = R * C * P   (R[i,2i]=1,R[i,2i+1]=-1 ; P[2j,j]=P[2j+1,j]=1)      -> channel 1
//   dxv = A * C * D   (A[i,2i]=A[i,2i+1]=1    ; D[2j,j]=1,D[2j+1,j]=-1)   -> channel 0
// Stage 1: T = C * {P|D} (256x128 f16 in LDS), Stage 2: {R|A} * T -> out (f32).
__global__ void __launch_bounds__(256) k_pool_wmma(const int* __restrict__ container,
                                                   float* __restrict__ out) {
  __shared__ _Float16 Ts[256 * 128];               // 64 KB of the 320 KB WGP LDS
  int b = blockIdx.x;
  int lane = threadIdx.x & 31;
  int w = threadIdx.x >> 5;                        // wave 0..7 (wave32)
  int rc = lane & 15;                              // A-row / B-col index within tile
  int kbase = (lane >> 4) << 3;                    // 0 or 8 (ISA 16-bit operand layout)
  const int* Cb = container + (b << 16);

  for (int pass = 0; pass < 2; ++pass) {
    float wodd_col = pass ? -1.0f : 1.0f;          // P vs D odd-row weight
    float wodd_row = pass ? 1.0f : -1.0f;          // A vs R odd-col weight
    // ---- stage 1: 128 tiles of T = C * {P|D}, 16 per wave, 1 WMMA each ----
    for (int q = 0; q < 16; ++q) {
      int t = w * 16 + q;
      int I = t >> 3, J = t & 7;                   // I: 16-row tile of C, J: 16-col tile of T
      v16h a, bb;
      const int* srcp = Cb + ((I << 4) + rc) * 256 + (J << 5);
#pragma unroll
      for (int j = 0; j < 16; ++j) {
        int k = kbase + j + ((j >= 8) ? 8 : 0);    // half j <-> K index (ISA 7.12.2)
        a[j]  = (_Float16)(float)srcp[k];
        bb[j] = (_Float16)(((k >> 1) == rc) ? ((k & 1) ? wodd_col : 1.0f) : 0.0f);
      }
      v8f acc = {};
      acc = __builtin_amdgcn_wmma_f32_16x16x32_f16(false, a, false, bb, (short)0, acc,
                                                   false, false);
      int mbase = (I << 4) + ((lane >> 4) << 3);
      int colg = (J << 4) + rc;
#pragma unroll
      for (int r = 0; r < 8; ++r) Ts[(mbase + r) * 128 + colg] = (_Float16)acc[r];
    }
    __syncthreads();
    // ---- stage 2: 64 tiles of {R|A} * T, 8 per wave, 1 WMMA each ----
    for (int q = 0; q < 8; ++q) {
      int t = w * 8 + q;
      int I2 = t >> 3, J2 = t & 7;
      v16h a, bb;
#pragma unroll
      for (int j = 0; j < 16; ++j) {
        int k = kbase + j + ((j >= 8) ? 8 : 0);
        a[j]  = (_Float16)(((k >> 1) == rc) ? ((k & 1) ? wodd_row : 1.0f) : 0.0f);
        bb[j] = Ts[((I2 << 5) + k) * 128 + (J2 << 4) + rc];
      }
      v8f acc = {};
      acc = __builtin_amdgcn_wmma_f32_16x16x32_f16(false, a, false, bb, (short)0, acc,
                                                   false, false);
      int ch = pass ? 0 : 1;                       // pass0 -> dyv (ch1), pass1 -> dxv (ch0)
      float* ob = out + ((long)b * 5 + ch) * HW4;
      int mbase = (I2 << 4) + ((lane >> 4) << 3);
      int colg = (J2 << 4) + rc;
#pragma unroll
      for (int r = 0; r < 8; ++r) ob[(mbase + r) * 128 + colg] = acc[r];
    }
    __syncthreads();
  }
}

// ------------------------------------------------------------------ channels 2,3,4
__global__ void k_finalize(const int* __restrict__ counter4, const float* __restrict__ timer4,
                           const float* __restrict__ seg, float* __restrict__ out) {
  int gid = blockIdx.x * blockDim.x + threadIdx.x;
  if (gid >= BB * HW4) return;
  int b = gid >> 14, rem = gid & (HW4 - 1);
  int yy = rem >> 7, xx = rem & 127;
  float c = (float)counter4[gid];
  float tm = timer4[gid] / (c == 0.0f ? 1.0f : c);
  float ch = seg[(b << 16) + (yy << 9) + (xx << 1)] - 16.0f;  // relu(segment - S)[::2,::2]
  ch = ch > 0.0f ? ch : 0.0f;
  float* ob = out + (long)b * 5 * HW4;
  ob[2 * HW4 + rem] = tm;
  ob[3 * HW4 + rem] = c;
  ob[4 * HW4 + rem] = ch;
}

// ------------------------------------------------------------------ launch
extern "C" void kernel_launch(void* const* d_in, const int* in_sizes, int n_in,
                              void* d_out, int out_size, void* d_ws, size_t ws_size,
                              hipStream_t stream) {
  const float* ev = (const float*)d_in[0];
  int n = in_sizes[0] / 5;
  char* ws = (char*)d_ws;
  unsigned* tmaxU   = (unsigned*)(ws + OFF_TMAX);
  unsigned* cnt     = (unsigned*)(ws + OFF_CNT);
  unsigned* xsum    = (unsigned*)(ws + OFF_XSUM);
  unsigned* ysum    = (unsigned*)(ws + OFF_YSUM);
  int4*     params  = (int4*)(ws + OFF_PARAMS);
  int*      counter4= (int*)(ws + OFF_COUNTER4);
  float*    timer4  = (float*)(ws + OFF_TIMER4);
  int*      container = (int*)(ws + OFF_CONTAINER);
  int*      combiner  = (int*)(ws + OFF_COMBINER);
  float*    segA    = (float*)(ws + OFF_SEG_A);
  float*    segB    = (float*)(ws + OFF_SEG_B);
  float*    out     = (float*)d_out;

  long n16 = (long)(ZERO_BYTES / 16u);
  k_zero<<<4096, 256, 0, stream>>>((uint4*)ws, n16);
  k_tmax<<<2048, 256, 0, stream>>>(ev, n, tmaxU);
  k_scatter<<<4096, 256, 0, stream>>>(ev, n, tmaxU, container, combiner,
                                      cnt, xsum, ysum, counter4, timer4);
  k_params<<<1, 32, 0, stream>>>(cnt, xsum, ysum, params);
  k_seg_init<<<(BB * HW) / 256, 256, 0, stream>>>(combiner, segA);
  float* srcb = segA;
  float* dstb = segB;
  for (int i = 1; i < SS; ++i) {
    k_combine<<<(BB * HW) / 256, 256, 0, stream>>>(params, combiner, srcb, dstb, i);
    float* tmp = srcb; srcb = dstb; dstb = tmp;
  }
  // final segment now in srcb
  k_pool_wmma<<<BB, 256, 0, stream>>>(container, out);
  k_finalize<<<(BB * HW4) / 256, 256, 0, stream>>>(counter4, timer4, srcb, out);
}
